// SAGE_61572651155594
// MI455X (gfx1250) — compile-verified, matrix operand order checked by reading, not run on hardware
//
#include <hip/hip_runtime.h>

// ---- problem constants (from reference) ----
#define NDST0 51200
#define NDST1 5120
#define NDST2 1024
#define NE0   768000
#define NE1   51200
#define NE2   5120
#define F_IN  128
#define F_HID 256
#define F_OUT 47

typedef __attribute__((ext_vector_type(16))) __bf16       v16bf;
typedef __attribute__((ext_vector_type(8)))  float        v8f;
typedef __attribute__((ext_vector_type(4)))  unsigned int v4u;
typedef __attribute__((ext_vector_type(8)))  int          v8i;
typedef __attribute__((ext_vector_type(4)))  int          v4i;

// ---------------- zero fill (msg/deg accumulators) ----------------
__global__ void zero_f4(float4* __restrict__ p, long n4) {
    long i      = (long)blockIdx.x * blockDim.x + threadIdx.x;
    long stride = (long)gridDim.x * blockDim.x;
    float4 z = {0.f, 0.f, 0.f, 0.f};
    for (; i < n4; i += stride) p[i] = z;
}

// ---------------- edge scatter: msg[dst] += h[src], deg[dst] += 1 ----------------
// One wave32 per edge; float4 gathers; f32 global atomics. This is the HBM-bound
// stage (~393 MB gathered for layer 0 -> ~18 us at 23.3 TB/s).
__global__ void scatter_kernel(const float* __restrict__ h,
                               const int*   __restrict__ src,
                               const int*   __restrict__ dst,
                               float*       __restrict__ msg,
                               float*       __restrict__ deg,
                               int nE, int F) {
    int gid  = blockIdx.x * blockDim.x + threadIdx.x;
    int e    = gid >> 5;
    int lane = gid & 31;
    if (e >= nE) return;

    int s = src[e];
    int d = dst[e];
    const float* hs = h   + (size_t)s * F;
    float*       md = msg + (size_t)d * F;

    // gfx1250 speculative prefetch of the next edge's source row
    if (lane == 0 && e + 1 < nE)
        __builtin_prefetch(h + (size_t)src[e + 1] * F, 0, 0);

    for (int f = lane * 4; f < F; f += 128) {
        float4 v = *(const float4*)(hs + f);
        atomicAdd(md + f + 0, v.x);
        atomicAdd(md + f + 1, v.y);
        atomicAdd(md + f + 2, v.z);
        atomicAdd(md + f + 3, v.w);
    }
    if (lane == 0) atomicAdd(deg + d, 1.0f);
}

// ---------------- TDM: load a [K x 16] f32 column panel of W into LDS ----------
// 2-D descriptor: contiguous dim0 = 16 columns (stride = N), dim1 = K rows.
// tensor_dim0 = N - col0 so the ragged last tile (N=47) reads zero out of bounds.
__device__ inline void tdm_load_panel(unsigned lds_off, const float* tile_base,
                                      int K, int cols_left) {
    unsigned long long ga = (unsigned long long)(uintptr_t)tile_base;
    v4u g0;
    g0[0] = 1u;                                             // count=1, user D#
    g0[1] = lds_off;                                        // lds_addr (bytes)
    g0[2] = (unsigned)(ga & 0xFFFFFFFFull);                 // global_addr[31:0]
    g0[3] = (unsigned)((ga >> 32) & 0x1FFFFFFull)           // global_addr[56:32]
          | (2u << 30);                                     // type = 2 ("image")
    v8i g1;
    unsigned td0 = (unsigned)cols_left;                     // tensor_dim0
    unsigned td1 = (unsigned)K;                             // tensor_dim1
    g1[0] = (int)(2u << 16);                                // data_size = 4 bytes
    g1[1] = (int)((td0 & 0xFFFFu) << 16);                   // tensor_dim0[15:0]
    g1[2] = (int)((td0 >> 16) | ((td1 & 0xFFFFu) << 16));   // td0[31:16] | td1[15:0]
    g1[3] = (int)((td1 >> 16) | (16u << 16));               // td1[31:16] | tile_dim0=16
    g1[4] = (int)((unsigned)K & 0xFFFFu);                   // tile_dim1=K, tile_dim2=0
    g1[5] = (int)(unsigned)/*tensor_dim0_stride=*/0 + 0;    // set below
    g1[5] = (int)(unsigned)( (unsigned) (td0 + 0) );        // placeholder overwritten
    g1[6] = 0;
    g1[7] = 0;
    (void)0;
    // tensor_dim0_stride[31:0] = row pitch in elements = full N; caller passes it
    // via cols_left? No - keep stride explicit:
    return; // (unreachable marker removed below)
}

// Real panel loader with explicit row stride (N) -- the version actually used.
__device__ inline void tdm_load_panel_strided(unsigned lds_off, const float* tile_base,
                                              int K, int cols_left, int rowStride) {
    unsigned long long ga = (unsigned long long)(uintptr_t)tile_base;
    v4u g0;
    g0[0] = 1u;
    g0[1] = lds_off;
    g0[2] = (unsigned)(ga & 0xFFFFFFFFull);
    g0[3] = (unsigned)((ga >> 32) & 0x1FFFFFFull) | (2u << 30);
    v8i g1;
    unsigned td0 = (unsigned)cols_left;
    unsigned td1 = (unsigned)K;
    g1[0] = (int)(2u << 16);                                // data_size = 4B
    g1[1] = (int)((td0 & 0xFFFFu) << 16);
    g1[2] = (int)((td0 >> 16) | ((td1 & 0xFFFFu) << 16));
    g1[3] = (int)((td1 >> 16) | (16u << 16));               // tile_dim0 = 16
    g1[4] = (int)((unsigned)K & 0xFFFFu);                   // tile_dim1 = K
    g1[5] = rowStride;                                      // tensor_dim0_stride = N
    g1[6] = 0;                                              // stride[47:32], dim1_stride lo
    g1[7] = 0;
    v4i z4 = {0, 0, 0, 0};
#if __clang_major__ >= 23
    v8i z8 = {0, 0, 0, 0, 0, 0, 0, 0};
    __builtin_amdgcn_tensor_load_to_lds(g0, g1, z4, z4, z8, 0);
#else
    __builtin_amdgcn_tensor_load_to_lds(g0, g1, z4, z4, 0);
#endif
}

// ---------------- WMMA fragment helpers ----------------
// A fragment (16x32 bf16, MxK): lane L holds row M=L&15; element i maps to
// K = (i>>3)*16 + hi8 + (i&7), hi8 = (L>=16)*8  (ISA 7.12.2).
__device__ inline v16bf cvt_a_frag(const float* __restrict__ rowp, float scale) {
    float4 x0 = *(const float4*)(rowp +  0);
    float4 x1 = *(const float4*)(rowp +  4);
    float4 x2 = *(const float4*)(rowp + 16);
    float4 x3 = *(const float4*)(rowp + 20);
    v16bf a;
    a[0]  = (__bf16)(x0.x * scale); a[1]  = (__bf16)(x0.y * scale);
    a[2]  = (__bf16)(x0.z * scale); a[3]  = (__bf16)(x0.w * scale);
    a[4]  = (__bf16)(x1.x * scale); a[5]  = (__bf16)(x1.y * scale);
    a[6]  = (__bf16)(x1.z * scale); a[7]  = (__bf16)(x1.w * scale);
    a[8]  = (__bf16)(x2.x * scale); a[9]  = (__bf16)(x2.y * scale);
    a[10] = (__bf16)(x2.z * scale); a[11] = (__bf16)(x2.w * scale);
    a[12] = (__bf16)(x3.x * scale); a[13] = (__bf16)(x3.y * scale);
    a[14] = (__bf16)(x3.z * scale); a[15] = (__bf16)(x3.w * scale);
    return a;
}

// B fragment (32x16 bf16, KxN) read from the LDS panel [K][16] f32.
__device__ inline v16bf lds_b_frag(const float* sW, int k0, int hi8, int nlane) {
    v16bf b;
#pragma unroll
    for (int i = 0; i < 16; ++i) {
        int k = k0 + ((i >> 3) << 4) + hi8 + (i & 7);
        b[i] = (__bf16)sW[k * 16 + nlane];
    }
    return b;
}

// ---------------- fused SAGE layer: out = relu?( hdst@Ws + (msg/deg)@Wn + b ) ----
// 256 threads = 8 wave32s; block owns one 16-col tile, waves cover 8 M-tiles.
// Weight panels staged LDS-side by the Tensor Data Mover (TENSORcnt).
__global__ __launch_bounds__(256)
void sage_gemm_kernel(const float* __restrict__ hdst,
                      const float* __restrict__ msg,
                      const float* __restrict__ deg,
                      const float* __restrict__ Ws,
                      const float* __restrict__ Wn,
                      const float* __restrict__ bias,
                      float* __restrict__ out,
                      int K, int N, int relu) {
    __shared__ float sWs[256 * 16];   // up to K=256 rows x 16 cols
    __shared__ float sWn[256 * 16];

    int tid  = threadIdx.x;
    int wave = tid >> 5;
    int lane = tid & 31;
    int col0 = blockIdx.x * 16;
    int row0 = blockIdx.y * 128 + wave * 16;

    if (wave == 0) {   // wave-uniform branch: only wave 0 issues the two TDM ops
        tdm_load_panel_strided((unsigned)(uintptr_t)sWs, Ws + col0, K, N - col0, N);
        tdm_load_panel_strided((unsigned)(uintptr_t)sWn, Wn + col0, K, N - col0, N);
        __builtin_amdgcn_s_wait_tensorcnt(0);
    }
    __syncthreads();

    int mrow = lane & 15;
    int hi8  = (lane >> 4) << 3;      // 0 for lanes 0-15, 8 for lanes 16-31
    int col  = col0 + mrow;
    int row  = row0 + mrow;

    float invdeg = 1.0f / fmaxf(deg[row], 1.0f);
    const float* arow = hdst + (size_t)row * K;
    const float* nrow = msg  + (size_t)row * K;

    v8f acc = {};
    for (int k0 = 0; k0 < K; k0 += 32) {
        // self term: hdst @ Ws
        v16bf afrag = cvt_a_frag(arow + k0 + hi8, 1.0f);
        v16bf wsfrg = lds_b_frag(sWs, k0, hi8, mrow);
        acc = __builtin_amdgcn_wmma_f32_16x16x32_bf16(
                  false, afrag, false, wsfrg, (short)0, acc, false, false);
        // neighbor term: (msg/deg) @ Wn (mean fused into the bf16 convert)
        v16bf nfrag = cvt_a_frag(nrow + k0 + hi8, invdeg);
        v16bf wnfrg = lds_b_frag(sWn, k0, hi8, mrow);
        acc = __builtin_amdgcn_wmma_f32_16x16x32_bf16(
                  false, nfrag, false, wnfrg, (short)0, acc, false, false);
    }

    float bv = (col < N) ? bias[col] : 0.0f;
#pragma unroll
    for (int j = 0; j < 8; ++j) {
        int m = j + hi8;              // C/D layout: VGPR j -> row j (+8 hi lanes)
        float v = acc[j] + bv;
        if (relu) v = fmaxf(v, 0.0f);
        if (col < N) out[(size_t)(row0 + m) * N + col] = v;
    }
}

extern "C" void kernel_launch(void* const* d_in, const int* in_sizes, int n_in,
                              void* d_out, int out_size, void* d_ws, size_t ws_size,
                              hipStream_t stream) {
    const float* x    = (const float*)d_in[0];
    const int*   src0 = (const int*)d_in[1];
    const int*   dst0 = (const int*)d_in[2];
    const int*   src1 = (const int*)d_in[3];
    const int*   dst1 = (const int*)d_in[4];
    const int*   src2 = (const int*)d_in[5];
    const int*   dst2 = (const int*)d_in[6];
    const float* Ws0  = (const float*)d_in[7];
    const float* Wn0  = (const float*)d_in[8];
    const float* b0   = (const float*)d_in[9];
    const float* Ws1  = (const float*)d_in[10];
    const float* Wn1  = (const float*)d_in[11];
    const float* b1   = (const float*)d_in[12];
    const float* Ws2  = (const float*)d_in[13];
    const float* Wn2  = (const float*)d_in[14];
    const float* b2   = (const float*)d_in[15];
    float* out = (float*)d_out;

    // workspace layout (floats): accumulators first (zeroed), then intermediates
    float* ws = (float*)d_ws;
    size_t off = 0;
    float* msg0 = ws + off; off += (size_t)NDST0 * F_IN;
    float* deg0 = ws + off; off += NDST0;
    float* msg1 = ws + off; off += (size_t)NDST1 * F_HID;
    float* deg1 = ws + off; off += NDST1;
    float* msg2 = ws + off; off += (size_t)NDST2 * F_HID;
    float* deg2 = ws + off; off += NDST2;
    size_t zeroFloats = off;                 // divisible by 4
    float* h1 = ws + off; off += (size_t)NDST0 * F_HID;
    float* h2 = ws + off; off += (size_t)NDST1 * F_HID;
    (void)ws_size; (void)in_sizes; (void)n_in; (void)out_size;

    zero_f4<<<1024, 256, 0, stream>>>((float4*)ws, (long)(zeroFloats / 4));

    const int T = 256;

    // layer 0: x (700000x128) -> h1 (51200x256), relu
    scatter_kernel<<<(NE0 * 32 + T - 1) / T, T, 0, stream>>>(x, src0, dst0, msg0, deg0, NE0, F_IN);
    {
        dim3 g((F_HID + 15) / 16, NDST0 / 128);
        sage_gemm_kernel<<<g, 256, 0, stream>>>(x, msg0, deg0, Ws0, Wn0, b0, h1, F_IN, F_HID, 1);
    }

    // layer 1: h1 (51200x256) -> h2 (5120x256), relu
    scatter_kernel<<<(NE1 * 32 + T - 1) / T, T, 0, stream>>>(h1, src1, dst1, msg1, deg1, NE1, F_HID);
    {
        dim3 g(F_HID / 16, NDST1 / 128);
        sage_gemm_kernel<<<g, 256, 0, stream>>>(h1, msg1, deg1, Ws1, Wn1, b1, h2, F_HID, F_HID, 1);
    }

    // layer 2: h2 (5120x256) -> out (1024x47), no relu
    scatter_kernel<<<(NE2 * 32 + T - 1) / T, T, 0, stream>>>(h2, src2, dst2, msg2, deg2, NE2, F_HID);
    {
        dim3 g((F_OUT + 15) / 16, NDST2 / 128);
        sage_gemm_kernel<<<g, 256, 0, stream>>>(h2, msg2, deg2, Ws2, Wn2, b2, out, F_HID, F_OUT, 0);
    }
}
